// AnomalyAwareSelfAttention_31490700214980
// MI455X (gfx1250) — matrix operand order repaired and checked by reading, not compiled, not run on hardware
//
#include <hip/hip_runtime.h>
#include <hip/hip_bf16.h>

// ---------------------------------------------------------------------------
// AnomalyAwareSelfAttention for MI455X (gfx1250, wave32, WMMA).
// All GEMMs use v_wmma_f32_16x16x32_f16 (f16 in, f32 accumulate).
// Block-cooperative B panel staged via global_load_async_to_lds_b128
// (double-buffered, ASYNCcnt-synchronized) -> 8x less L2 traffic for B.
// Wave tile 32(M) x 64(N); block tile 256(M) x 64(N) with 8 waves.
// ---------------------------------------------------------------------------

typedef __attribute__((ext_vector_type(16))) _Float16 v16h;
typedef __attribute__((ext_vector_type(8)))  _Float16 v8h;
typedef __attribute__((ext_vector_type(4)))  _Float16 v4h;
typedef __attribute__((ext_vector_type(8)))  float    v8f;
typedef __attribute__((address_space(3)))    char     lds_char_t;

#define HID   1024
#define BATCH 8
#define SEQ   2048
#define NTOK  (BATCH * SEQ)   // 16384 tokens

// ---- WMMA A-operand loader (contiguous k order, same as B) ----------------------
// A-matrix 16x32 f16: lane L(<16): M=L, halfs = K[0..15];
//                     lane L+16  : M=L, halfs = K[16..31].
__device__ __forceinline__ v16h load_a16x32(const _Float16* __restrict__ X, int ld,
                                            int row_base, int k_base, int lane) {
  const int row = row_base + (lane & 15);
  const int k0  = k_base + ((lane >> 4) << 4);
  const _Float16* p = X + (long long)row * ld + k0;
  v8h lo = *(const v8h*)p;          // K k0..k0+7
  v8h hi = *(const v8h*)(p + 8);    // K k0+8..k0+15
  return __builtin_shufflevector(lo, hi, 0,1,2,3,4,5,6,7,8,9,10,11,12,13,14,15);
}

// ---------------------------------------------------------------------------
// Kernel 1: per-token L2 norm; write scaled f16 activations + fp32 norms.
// ---------------------------------------------------------------------------
__global__ void prep_norm_scale(const float* __restrict__ h,
                                _Float16* __restrict__ xh,
                                float* __restrict__ norms) {
  const long long row = blockIdx.x;
  const float4 v = ((const float4*)(h + row * HID))[threadIdx.x];
  float ss = v.x * v.x + v.y * v.y + v.z * v.z + v.w * v.w;
#pragma unroll
  for (int off = 16; off > 0; off >>= 1) ss += __shfl_xor(ss, off, 32);
  __shared__ float sred[8];
  const int wid = threadIdx.x >> 5, lane = threadIdx.x & 31;
  if (lane == 0) sred[wid] = ss;
  __syncthreads();
  float tot = 0.f;
#pragma unroll
  for (int i = 0; i < 8; ++i) tot += sred[i];
  const float nrm = sqrtf(tot);
  const float inv = 1.f / (nrm + 1e-9f);
  if (threadIdx.x == 0) norms[row] = nrm;
  v4h o;
  o[0] = (_Float16)(v.x * inv); o[1] = (_Float16)(v.y * inv);
  o[2] = (_Float16)(v.z * inv); o[3] = (_Float16)(v.w * inv);
  ((v4h*)(xh + row * HID))[threadIdx.x] = o;
}

// ---------------------------------------------------------------------------
// Kernel 2: f16 conversion of Wq, Wv and transposed anomaly matrix.
// at[o*H + h] = A[h*H + o]  so tq = q @ A becomes q @ At^T (same GEMM form).
// ---------------------------------------------------------------------------
__global__ void convert_weights(const float* __restrict__ Wq,
                                const float* __restrict__ Wv,
                                const float* __restrict__ Am,
                                _Float16* __restrict__ wqh,
                                _Float16* __restrict__ wvh,
                                _Float16* __restrict__ ath) {
  const int stride = gridDim.x * blockDim.x;
  for (int i = blockIdx.x * blockDim.x + threadIdx.x; i < HID * HID; i += stride) {
    wqh[i] = (_Float16)Wq[i];
    wvh[i] = (_Float16)Wv[i];
    const int o = i >> 10, hh = i & (HID - 1);
    ath[i] = (_Float16)Am[hh * HID + o];
  }
}

// ---------------------------------------------------------------------------
// Block-cooperative GEMM core with async B staging.
// B panel slice per K-step: 64 cols x 32 K halfs = 4 KB, layout
// [col][64B k-slice]; each of 256 threads DMAs one 16B chunk into LDS via
// global_load_async_to_lds_b128. Double-buffered (8 KB), ASYNCcnt-sync'd.
// Per-lane B fragment read (contiguous k order, same as A): lane n(<16): N=n, K=0..15;
// lane n+16: N=n, K=16..31 -> 32B contiguous at [col*64 + (lane>>4)*32].
// ---------------------------------------------------------------------------
__device__ __forceinline__ void gemm_core_async(
    const _Float16* __restrict__ A, int lda, int s_base,
    const _Float16* __restrict__ Bt, int ldb, int nb, int K,
    char* smem, v8f (&acc)[2][4]) {
  const int tid  = threadIdx.x;
  const int lane = tid & 31;
  // producer mapping: thread -> (col = tid/4, byte = (tid%4)*16) of 4KB panel
  const int pcol  = tid >> 2;
  const int pbyte = (tid & 3) << 4;
  const char* gp = (const char*)(Bt + (size_t)(nb + pcol) * ldb) + pbyte;
  const uint32_t lds0  = (uint32_t)(unsigned long long)(lds_char_t*)smem;
  const uint32_t myoff = lds0 + (uint32_t)tid * 16u;

  // prologue: stage k-slice 0 into buffer 0
  asm volatile("global_load_async_to_lds_b128 %0, %1, off"
               :: "v"(myoff), "v"(gp) : "memory");
  int stage = 0;
  for (int kk = 0; kk < K; kk += 32) {
    if (kk + 32 < K) {
      const char* srcn = gp + (size_t)(kk + 32) * 2;
      const uint32_t dstn = myoff + (uint32_t)((stage ^ 1) << 12);
      asm volatile("global_load_async_to_lds_b128 %0, %1, off"
                   :: "v"(dstn), "v"(srcn) : "memory");
      asm volatile("s_wait_asynccnt 0x1" ::: "memory");   // current slice done
    } else {
      asm volatile("s_wait_asynccnt 0x0" ::: "memory");
    }
    __syncthreads();   // publish buffer[stage] to all waves

    const v16h a0 = load_a16x32(A, lda, s_base,      kk, lane);
    const v16h a1 = load_a16x32(A, lda, s_base + 16, kk, lane);
    const char* bb = smem + (stage << 12) + ((lane & 15) << 6) + ((lane >> 4) << 5);
#pragma unroll
    for (int t = 0; t < 4; ++t) {
      const v16h b = *(const v16h*)(bb + (t << 10));
      acc[0][t] = __builtin_amdgcn_wmma_f32_16x16x32_f16(
          false, a0, false, b, (short)0, acc[0][t], false, false);
      acc[1][t] = __builtin_amdgcn_wmma_f32_16x16x32_f16(
          false, a1, false, b, (short)0, acc[1][t], false, false);
    }
    __syncthreads();   // all consumed before buffer[stage] is overwritten
    stage ^= 1;
  }
}

#define WMMA_ACC_INIT(acc)                                          \
  const v8f zero_ = {0.f, 0.f, 0.f, 0.f, 0.f, 0.f, 0.f, 0.f};      \
  v8f acc[2][4] = {{zero_, zero_, zero_, zero_},                    \
                   {zero_, zero_, zero_, zero_}};

// ---------------------------------------------------------------------------
// Generic GEMM: C[m,n] = (sum_k A[m,k] * Bt[n,k] + bias[n]) * scale, f16 out.
// Block tile 256(M) x 64(N), 8 waves of 32(M) each. Batched via blockIdx.y.
// ---------------------------------------------------------------------------
template <bool HAS_BIAS>
__global__ void wmma_gemm_rowf16(const _Float16* __restrict__ A, int lda, long long strideA,
                                 const _Float16* __restrict__ Bt, int ldb, long long strideB,
                                 _Float16* __restrict__ C, int ldc, long long strideC,
                                 const float* __restrict__ bias,
                                 int N, int K, float scale) {
  __shared__ __align__(128) char smemB[8192];
  const int lane = threadIdx.x & 31;
  const int wid  = threadIdx.x >> 5;
  const int n64  = N >> 6;
  const int s_base = (blockIdx.x / n64) * 256 + wid * 32;
  const int nb     = (blockIdx.x % n64) << 6;
  A  += (long long)blockIdx.y * strideA;
  Bt += (long long)blockIdx.y * strideB;
  C  += (long long)blockIdx.y * strideC;

  WMMA_ACC_INIT(acc);
  gemm_core_async(A, lda, s_base, Bt, ldb, nb, K, smemB, acc);

  // D layout: lane n holds column N=n; VGPR r holds row M=r (+8 for lanes 16-31).
  const int col0  = nb + (lane & 15);
  const int rbase = s_base + ((lane >> 4) << 3);
#pragma unroll
  for (int t = 0; t < 4; ++t) {
    float badd = 0.f;
    if (HAS_BIAS) badd = bias[col0 + 16 * t];
#pragma unroll
    for (int m = 0; m < 2; ++m)
#pragma unroll
      for (int r = 0; r < 8; ++r) {
        const float val = (acc[m][t][r] + badd) * scale;
        C[(long long)(rbase + 16 * m + r) * ldc + col0 + 16 * t] = (_Float16)val;
      }
  }
}

// ---------------------------------------------------------------------------
// V projection with TRANSPOSED output: Vt[b][o][s] = scaled@Wv^T + bv.
// D layout gives each lane 8 consecutive s-values -> packed 16B v8h store.
// ---------------------------------------------------------------------------
__global__ void wmma_gemm_v_trans(const _Float16* __restrict__ X,
                                  const _Float16* __restrict__ Wvh,
                                  const float* __restrict__ bias,
                                  _Float16* __restrict__ Vt) {
  __shared__ __align__(128) char smemB[8192];
  const int lane = threadIdx.x & 31;
  const int wid  = threadIdx.x >> 5;
  const int n64  = HID >> 6;                 // 16
  const int s_base = (blockIdx.x / n64) * 256 + wid * 32;
  const int nb     = (blockIdx.x % n64) << 6;

  WMMA_ACC_INIT(acc);
  gemm_core_async(X, HID, s_base, Wvh, HID, nb, HID, smemB, acc);

  const int col0 = nb + (lane & 15);
  const int bb   = s_base >> 11;                         // batch (S = 2048)
  const int sl   = (s_base & (SEQ - 1)) + ((lane >> 4) << 3);
#pragma unroll
  for (int t = 0; t < 4; ++t) {
    const float badd = bias[col0 + 16 * t];
    _Float16* base = Vt + ((size_t)bb * HID + col0 + 16 * t) * SEQ + sl;
#pragma unroll
    for (int m = 0; m < 2; ++m) {
      v8h o;
#pragma unroll
      for (int r = 0; r < 8; ++r) o[r] = (_Float16)(acc[m][t][r] + badd);
      *(v8h*)(base + 16 * m) = o;
    }
  }
}

// ---------------------------------------------------------------------------
// Row softmax over S=2048 f16 scores, in place. One block per row.
// ---------------------------------------------------------------------------
__global__ void softmax_rows(_Float16* __restrict__ scores) {
  _Float16* p = scores + (long long)blockIdx.x * SEQ;
  v8h v = *(const v8h*)(p + threadIdx.x * 8);
  float f[8];
  float m = -1e30f;
#pragma unroll
  for (int i = 0; i < 8; ++i) { f[i] = (float)v[i]; m = fmaxf(m, f[i]); }
#pragma unroll
  for (int off = 16; off > 0; off >>= 1) m = fmaxf(m, __shfl_xor(m, off, 32));
  __shared__ float sred[8];
  const int wid = threadIdx.x >> 5, lane = threadIdx.x & 31;
  if (lane == 0) sred[wid] = m;
  __syncthreads();
  float gm = -1e30f;
#pragma unroll
  for (int i = 0; i < 8; ++i) gm = fmaxf(gm, sred[i]);
  __syncthreads();
  float s = 0.f;
#pragma unroll
  for (int i = 0; i < 8; ++i) { f[i] = __expf(f[i] - gm); s += f[i]; }
#pragma unroll
  for (int off = 16; off > 0; off >>= 1) s += __shfl_xor(s, off, 32);
  if (lane == 0) sred[wid] = s;
  __syncthreads();
  float gs = 0.f;
#pragma unroll
  for (int i = 0; i < 8; ++i) gs += sred[i];
  const float inv = 1.f / gs;
  v8h o;
#pragma unroll
  for (int i = 0; i < 8; ++i) o[i] = (_Float16)(f[i] * inv);
  *(v8h*)(p + threadIdx.x * 8) = o;
}

// ---------------------------------------------------------------------------
// ctx GEMM: out[b][s][h] = (sum_t P[b][s][t] * Vt[b][h][t]) * norms[b*S+s].
// fp32 output. Batched via blockIdx.y.
// ---------------------------------------------------------------------------
__global__ void wmma_gemm_ctx(const _Float16* __restrict__ P,
                              const _Float16* __restrict__ Vt,
                              const float* __restrict__ norms,
                              float* __restrict__ out) {
  __shared__ __align__(128) char smemB[8192];
  const int lane = threadIdx.x & 31;
  const int wid  = threadIdx.x >> 5;
  const int n64  = HID >> 6;                 // 16
  const int s_base = (blockIdx.x / n64) * 256 + wid * 32;
  const int nb     = (blockIdx.x % n64) << 6;
  const long long b = blockIdx.y;
  const _Float16* A  = P  + b * (long long)SEQ * SEQ;
  const _Float16* Bt = Vt + b * (long long)HID * SEQ;
  float* C           = out + b * (long long)SEQ * HID;
  const float* nb_norms = norms + b * SEQ;

  WMMA_ACC_INIT(acc);
  gemm_core_async(A, SEQ, s_base, Bt, SEQ, nb, SEQ, smemB, acc);

  const int col0  = nb + (lane & 15);
  const int rbase = s_base + ((lane >> 4) << 3);
  float nr[2][8];
#pragma unroll
  for (int m = 0; m < 2; ++m)
#pragma unroll
    for (int r = 0; r < 8; ++r) nr[m][r] = nb_norms[rbase + 16 * m + r];
#pragma unroll
  for (int t = 0; t < 4; ++t)
#pragma unroll
    for (int m = 0; m < 2; ++m)
#pragma unroll
      for (int r = 0; r < 8; ++r)
        C[(long long)(rbase + 16 * m + r) * HID + col0 + 16 * t] =
            acc[m][t][r] * nr[m][r];
}

// ---------------------------------------------------------------------------
// Launcher
// ---------------------------------------------------------------------------
extern "C" void kernel_launch(void* const* d_in, const int* in_sizes, int n_in,
                              void* d_out, int out_size, void* d_ws, size_t ws_size,
                              hipStream_t stream) {
  (void)in_sizes; (void)n_in; (void)out_size; (void)ws_size;
  const float* hs = (const float*)d_in[0];
  const float* Wq = (const float*)d_in[1];
  const float* bq = (const float*)d_in[2];
  const float* Wv = (const float*)d_in[3];
  const float* bv = (const float*)d_in[4];
  const float* Am = (const float*)d_in[5];
  float* out = (float*)d_out;

  char* ws = (char*)d_ws;
  size_t off = 0;
  auto carve = [&](size_t bytes) -> void* {
    off = (off + 255) & ~(size_t)255;
    void* p = ws + off;
    off += bytes;
    return p;
  };
  _Float16* xh    = (_Float16*)carve((size_t)NTOK * HID * 2);       // 32 MB
  _Float16* qh    = (_Float16*)carve((size_t)NTOK * HID * 2);       // 32 MB
  _Float16* tqh   = (_Float16*)carve((size_t)NTOK * HID * 2);       // 32 MB
  _Float16* vt    = (_Float16*)carve((size_t)NTOK * HID * 2);       // 32 MB (as [B][H][S])
  _Float16* probs = (_Float16*)carve((size_t)BATCH * SEQ * SEQ * 2);// 64 MB
  float*    norms = (float*)   carve((size_t)NTOK * 4);             // 64 KB
  _Float16* wqh   = (_Float16*)carve((size_t)HID * HID * 2);        // 2 MB
  _Float16* wvh   = (_Float16*)carve((size_t)HID * HID * 2);        // 2 MB
  _Float16* ath   = (_Float16*)carve((size_t)HID * HID * 2);        // 2 MB

  // 1) per-token L2 norm + f16 scaled activations
  prep_norm_scale<<<NTOK, 256, 0, stream>>>(hs, xh, norms);
  // 2) weight conversion + anomaly-matrix transpose to f16
  convert_weights<<<1024, 256, 0, stream>>>(Wq, Wv, Am, wqh, wvh, ath);

  const int blocks_full = (NTOK / 256) * (HID / 64);   // 1024 blocks (8 waves each)
  // 3) Q = scaled @ Wq^T + bq   (f16 row-major [NTOK, H])
  wmma_gemm_rowf16<true><<<blocks_full, 256, 0, stream>>>(
      xh, HID, 0, wqh, HID, 0, qh, HID, 0, bq, HID, HID, 1.0f);
  // 4) Vt = (scaled @ Wv^T + bv)^T  (f16 [B][H][S])
  wmma_gemm_v_trans<<<blocks_full, 256, 0, stream>>>(xh, wvh, bv, vt);
  // 5) TQ = Q @ A  (via At, f16 row-major)
  wmma_gemm_rowf16<false><<<blocks_full, 256, 0, stream>>>(
      qh, HID, 0, ath, HID, 0, tqh, HID, 0, nullptr, HID, HID, 1.0f);
  // 6) scores[b] = (Q[b] @ TQ[b]^T) / 32  (f16 [B][S][S])
  {
    dim3 grid((SEQ / 256) * (SEQ / 64), BATCH);        // 256 x 8
    wmma_gemm_rowf16<false><<<grid, 256, 0, stream>>>(
        qh, HID, (long long)SEQ * HID, tqh, HID, (long long)SEQ * HID,
        probs, SEQ, (long long)SEQ * SEQ, nullptr, SEQ, HID, 0.03125f);
  }
  // 7) softmax over rows, in place
  softmax_rows<<<BATCH * SEQ, 256, 0, stream>>>(probs);
  // 8) out[b] = (P[b] @ V[b]) * input_norm
  {
    dim3 grid((SEQ / 256) * (HID / 64), BATCH);        // 128 x 8
    wmma_gemm_ctx<<<grid, 256, 0, stream>>>(probs, vt, norms, out);
  }
}
